// DroneRelationModel_8504035246713
// MI455X (gfx1250) — matrix operand order repaired and checked by reading, not compiled, not run on hardware
//
#include <hip/hip_runtime.h>

typedef __attribute__((ext_vector_type(16))) _Float16 v16h;
typedef __attribute__((ext_vector_type(8)))  float    v8f;

#define NN 4096
#define HH 128
#define NHEADS 4
#define DH 32
#define NPAIR 262144

// ---------- WMMA fragment loaders (CDNA5 16x16x32 f16 layouts) ----------
// A matrix 16x32 (MxK), element (m,k) at base[m*ld + k]
static __device__ inline v16h load_a16x32(const _Float16* base, int ld) {
  int lane = threadIdx.x & 31;
  int m = lane & 15;
  int koff = (lane >> 4) << 3;      // lanes 16-31 hold K+8
  v16h a;
#pragma unroll
  for (int i = 0; i < 16; ++i) {
    int k = (i & 7) + ((i >> 3) << 4) + koff;
    a[i] = base[m * ld + k];
  }
  return a;
}
// B matrix 32x16 (KxN) given transposed: element (n,k) at base[n*ld + k]
static __device__ inline v16h load_bt16x32(const _Float16* base, int ld) {
  int lane = threadIdx.x & 31;
  int n = lane & 15;
  int koff = (lane >> 4) << 3;      // lanes 16-31 hold K+8
  v16h b;
#pragma unroll
  for (int i = 0; i < 16; ++i) b[i] = base[n * ld + koff + (i & 7) + ((i >> 3) << 4)];
  return b;
}

static __device__ inline v8f wmma_f16(v16h a, v16h b, v8f c) {
  return __builtin_amdgcn_wmma_f32_16x16x32_f16(false, a, false, b, (short)0, c, false, false);
}

// 16x16 tile GEMM over K=128 from LDS (A: 16xK, Bt: 16 cols x K), lda padded
static __device__ inline v8f mm16x16x128(const _Float16* sA, const _Float16* sBt, int ld) {
  v8f c = {};
#pragma unroll
  for (int kk = 0; kk < 128; kk += 32) {
    v16h a = load_a16x32(sA + kk, ld);
    v16h b = load_bt16x32(sBt + kk, ld);
    c = wmma_f16(a, b, c);
  }
  return c;
}

#define LDA 136  // padded LDS row stride (halves): 272B = 17*16B (b128-aligned, bank stride 68%64=4 -> conflict free)

// async copy of 16B chunks from global f16 into LDS (gfx1250 async-LDS path, ASYNCcnt)
static __device__ inline void async_b128_to_lds(_Float16* lds_dst, const _Float16* gsrc) {
  unsigned lds_off = (unsigned)(size_t)(void*)lds_dst;  // low 32 bits of flat shared ptr = LDS offset
  asm volatile("global_load_async_to_lds_b128 %0, %1, off"
               :: "v"(lds_off), "v"(gsrc) : "memory");
}
static __device__ inline void wait_async0() {
  asm volatile("s_wait_asynccnt 0" ::: "memory");
}

// ---------------- Kernel 1: QKV projection ----------------
// qkv = x(4096x128) @ Wqkv(384x128)^T + b; scatter to f16 Q/K [h][n][d] and Vt [h][d][n]
__global__ void qkv_kernel(const float* __restrict__ x, const float* __restrict__ w,
                           const float* __restrict__ bias,
                           _Float16* __restrict__ Q, _Float16* __restrict__ K,
                           _Float16* __restrict__ Vt) {
  __shared__ __align__(16) _Float16 sX[16 * LDA];
  __shared__ __align__(16) _Float16 sW[64 * LDA];
  int rb = blockIdx.y, cb = blockIdx.x;
  int tid = threadIdx.x;
  for (int i = tid; i < 16 * 128; i += 128) {
    int r = i >> 7, c = i & 127;
    sX[r * LDA + c] = (_Float16)x[(rb * 16 + r) * 128 + c];
  }
  for (int i = tid; i < 64 * 128; i += 128) {
    int r = i >> 7, c = i & 127;
    sW[r * LDA + c] = (_Float16)w[(cb * 64 + r) * 128 + c];
  }
  __syncthreads();
  int wv = tid >> 5, lane = tid & 31;
  v8f c = mm16x16x128(sX, sW + wv * 16 * LDA, LDA);
  int n = lane & 15, moff = (lane >> 4) << 3;
  int col = cb * 64 + wv * 16 + n;
  float bcol = bias[col];
#pragma unroll
  for (int e = 0; e < 8; ++e) {
    int row = rb * 16 + e + moff;
    _Float16 hv = (_Float16)(c[e] + bcol);
    if (col < 128) {
      Q[(((col >> 5) * NN) + row) * DH + (col & 31)] = hv;
    } else if (col < 256) {
      int j = col - 128;
      K[(((j >> 5) * NN) + row) * DH + (j & 31)] = hv;
    } else {
      int j = col - 256;  // V stored transposed: Vt[h][d][n]
      Vt[((size_t)(j >> 5) * DH + (j & 31)) * NN + row] = hv;
    }
  }
}

// ---------------- Kernel 2: attention (per head, 16 q-rows / block) ----------------
#define SROW 4104  // 4096 + 8 pad (halves)
__global__ void attn_kernel(const _Float16* __restrict__ Q, const _Float16* __restrict__ K,
                            const _Float16* __restrict__ Vt, _Float16* __restrict__ ctx) {
  extern __shared__ char smem[];
  _Float16* Sbuf = (_Float16*)smem;                              // 16 * SROW halves
  float* red    = (float*)(smem + 16 * SROW * sizeof(_Float16)); // 128
  float* rowmax = red + 128;                                     // 16
  float* rowsum = rowmax + 16;                                   // 16
  float* Opart  = rowsum + 16;                                   // 4*16*32

  const int qb = blockIdx.x, h = blockIdx.y;
  const int q0 = qb * 16;
  const int tid = threadIdx.x, wv = tid >> 5, lane = tid & 31;
  const _Float16* Qh  = Q  + (size_t)h * NN * DH;
  const _Float16* Kh  = K  + (size_t)h * NN * DH;
  const _Float16* Vth = Vt + (size_t)h * DH * NN;
  const float scale = 0.17677669529663687f;  // 1/sqrt(32)
  const int n = lane & 15, moff = (lane >> 4) << 3;

  v16h qa = load_a16x32(Qh + q0 * DH, DH);
  // scores -> Sbuf (f16)
  for (int t = wv; t < NN / 16; t += 4) {
    int kb = t * 16;
    __builtin_prefetch(Kh + (kb + 256) * DH, 0, 0);
    v16h kf = load_bt16x32(Kh + kb * DH, DH);  // Bt[n][k] = K[key][d]
    v8f s = {};
    s = wmma_f16(qa, kf, s);
#pragma unroll
    for (int e = 0; e < 8; ++e)
      Sbuf[(e + moff) * SROW + kb + n] = (_Float16)(s[e] * scale);
  }
  __syncthreads();
  // row max (16 rows x 8 partitions)
  {
    int row = tid & 15, part = tid >> 4;
    float m = -3.0e38f;
    for (int c = part * 512; c < part * 512 + 512; ++c)
      m = fmaxf(m, (float)Sbuf[row * SROW + c]);
    red[tid] = m;
  }
  __syncthreads();
  if (tid < 16) {
    float m = red[tid];
    for (int p = 1; p < 8; ++p) m = fmaxf(m, red[tid + p * 16]);
    rowmax[tid] = m;
  }
  __syncthreads();
  // exp in place + row sum
  {
    int row = tid & 15, part = tid >> 4;
    float mx = rowmax[row], sum = 0.f;
    for (int c = part * 512; c < part * 512 + 512; ++c) {
      float e = __expf((float)Sbuf[row * SROW + c] - mx);
      Sbuf[row * SROW + c] = (_Float16)e;
      sum += e;
    }
    red[tid] = sum;
  }
  __syncthreads();
  if (tid < 16) {
    float s = 0.f;
    for (int p = 0; p < 8; ++p) s += red[tid + p * 16];
    rowsum[tid] = s;
  }
  __syncthreads();
  // O = P @ V, per-wave key partition, f32 accumulate; B-fragments from transposed V
  v8f o0 = {}, o1 = {};
  for (int kb = wv * 32; kb < NN; kb += 128) {
    __builtin_prefetch(Vth + kb + 128, 0, 0);
    v16h pa = load_a16x32(Sbuf + kb, SROW);
    v16h v0 = load_bt16x32(Vth + kb, NN);             // d = 0..15
    o0 = wmma_f16(pa, v0, o0);
    v16h v1 = load_bt16x32(Vth + 16 * NN + kb, NN);   // d = 16..31
    o1 = wmma_f16(pa, v1, o1);
  }
#pragma unroll
  for (int e = 0; e < 8; ++e) {
    Opart[(wv * 16 + e + moff) * 32 + n]      = o0[e];
    Opart[(wv * 16 + e + moff) * 32 + 16 + n] = o1[e];
  }
  __syncthreads();
  for (int idx = tid; idx < 512; idx += 128) {
    int m = idx >> 5, d = idx & 31;
    float v = 0.f;
    for (int p = 0; p < 4; ++p) v += Opart[(p * 16 + m) * 32 + d];
    v /= rowsum[m];
    ctx[(q0 + m) * HH + h * DH + d] = (_Float16)v;
  }
}

// ---------------- Kernel 3: out projection ----------------
__global__ void outproj_kernel(const _Float16* __restrict__ ctx, const float* __restrict__ w,
                               const float* __restrict__ bias, _Float16* __restrict__ ne) {
  __shared__ __align__(16) _Float16 sX[16 * LDA];
  __shared__ __align__(16) _Float16 sW[64 * LDA];
  int rb = blockIdx.y, cb = blockIdx.x;
  int tid = threadIdx.x;
  // X tile is already f16 in global: stage via async global->LDS (b128 chunks)
  for (int i = tid; i < 256; i += 128) {
    int r = i >> 4, c = (i & 15) * 8;
    async_b128_to_lds(&sX[r * LDA + c], ctx + (rb * 16 + r) * 128 + c);
  }
  for (int i = tid; i < 64 * 128; i += 128) {
    int r = i >> 7, c = i & 127;
    sW[r * LDA + c] = (_Float16)w[(cb * 64 + r) * 128 + c];
  }
  wait_async0();
  __syncthreads();
  int wv = tid >> 5, lane = tid & 31;
  v8f c = mm16x16x128(sX, sW + wv * 16 * LDA, LDA);
  int n = lane & 15, moff = (lane >> 4) << 3;
  int col = cb * 64 + wv * 16 + n;
  float bcol = bias[col];
#pragma unroll
  for (int e = 0; e < 8; ++e) {
    int row = rb * 16 + e + moff;
    ne[row * HH + col] = (_Float16)(c[e] + bcol);
  }
}

// ---------------- Kernel 4: pair tables A = ne@W1a^T + b1, B = ne@W1b^T ----------------
__global__ void tables_kernel(const _Float16* __restrict__ ne, const float* __restrict__ w1,
                              const float* __restrict__ b1, float* __restrict__ Atab,
                              float* __restrict__ Btab) {
  __shared__ __align__(16) _Float16 sX[16 * LDA];
  __shared__ __align__(16) _Float16 sW[64 * LDA];
  int rb = blockIdx.y, cb = blockIdx.x, z = blockIdx.z;
  int tid = threadIdx.x;
  for (int i = tid; i < 256; i += 128) {
    int r = i >> 4, c = (i & 15) * 8;
    async_b128_to_lds(&sX[r * LDA + c], ne + (rb * 16 + r) * 128 + c);
  }
  for (int i = tid; i < 64 * 128; i += 128) {
    int r = i >> 7, c = i & 127;
    sW[r * LDA + c] = (_Float16)w1[(cb * 64 + r) * 256 + z * 128 + c];
  }
  wait_async0();
  __syncthreads();
  int wv = tid >> 5, lane = tid & 31;
  v8f c = mm16x16x128(sX, sW + wv * 16 * LDA, LDA);
  int n = lane & 15, moff = (lane >> 4) << 3;
  int col = cb * 64 + wv * 16 + n;
  float bcol = (z == 0) ? b1[col] : 0.f;
  float* dst = (z == 0) ? Atab : Btab;
#pragma unroll
  for (int e = 0; e < 8; ++e) {
    int row = rb * 16 + e + moff;
    dst[row * HH + col] = c[e] + bcol;
  }
}

// ---------------- Kernel 5: per-pair epilogue (one wave32 per pair) ----------------
__global__ void pair_kernel(const long long* __restrict__ rel, const float* __restrict__ Atab,
                            const float* __restrict__ Btab, const float* __restrict__ w2,
                            const float* __restrict__ b2, float* __restrict__ out) {
  int gw = (int)((blockIdx.x * (unsigned)blockDim.x + threadIdx.x) >> 5);
  int lane = threadIdx.x & 31;
  long long s = rel[2 * (long long)gw];
  long long d = rel[2 * (long long)gw + 1];
  const float* arow = Atab + s * HH;
  const float* brow = Btab + d * HH;
  float acc = 0.f;
#pragma unroll
  for (int i = 0; i < 4; ++i) {
    int j = lane + i * 32;
    float t = arow[j] + brow[j];
    t = t > 0.f ? t : 0.f;
    acc += t * w2[j];
  }
#pragma unroll
  for (int off = 16; off; off >>= 1) acc += __shfl_down(acc, off, 32);
  if (lane == 0) out[gw] = 1.f / (1.f + __expf(-(acc + b2[0])));
}

extern "C" void kernel_launch(void* const* d_in, const int* in_sizes, int n_in,
                              void* d_out, int out_size, void* d_ws, size_t ws_size,
                              hipStream_t stream) {
  const float* x        = (const float*)d_in[1];   // context_embeddings (4096x128)
  const long long* rel  = (const long long*)d_in[2];
  const float* wqkv     = (const float*)d_in[3];
  const float* bqkv     = (const float*)d_in[4];
  const float* wo       = (const float*)d_in[5];
  const float* bo       = (const float*)d_in[6];
  const float* w1       = (const float*)d_in[7];
  const float* b1       = (const float*)d_in[8];
  const float* w2       = (const float*)d_in[9];
  const float* b2       = (const float*)d_in[10];
  float* out = (float*)d_out;

  char* ws = (char*)d_ws;
  _Float16* Q16 = (_Float16*)ws;                  // HEADS*N*DH
  _Float16* K16 = Q16 + NHEADS * NN * DH;
  _Float16* Vt16 = K16 + NHEADS * NN * DH;        // transposed V: [h][d][n]
  _Float16* CTX = Vt16 + NHEADS * NN * DH;        // N*H
  _Float16* NE  = CTX + NN * HH;                  // N*H
  float* Atab = (float*)(NE + NN * HH);           // N*H f32
  float* Btab = Atab + NN * HH;

  qkv_kernel<<<dim3(3 * HH / 64, NN / 16), 128, 0, stream>>>(x, wqkv, bqkv, Q16, K16, Vt16);

  size_t smem = (size_t)16 * SROW * sizeof(_Float16) + (128 + 16 + 16 + 4 * 16 * 32) * sizeof(float);
  attn_kernel<<<dim3(NN / 16, NHEADS), 128, smem, stream>>>(Q16, K16, Vt16, CTX);

  outproj_kernel<<<dim3(HH / 64, NN / 16), 128, 0, stream>>>(CTX, wo, bo, NE);
  tables_kernel<<<dim3(HH / 64, NN / 16, 2), 128, 0, stream>>>(NE, w1, b1, Atab, Btab);
  pair_kernel<<<NPAIR / 8, 256, 0, stream>>>(rel, Atab, Btab, w2, b2, out);
}
